// aformer_39376260169836
// MI455X (gfx1250) — matrix operand order repaired and checked by reading, not compiled, not run on hardware
//
#include <hip/hip_runtime.h>
#include <hip/hip_bf16.h>
#include <math.h>

typedef __attribute__((ext_vector_type(16))) _Float16 v16h;
typedef __attribute__((ext_vector_type(8)))  float    v8f;

// ---------------------------------------------------------------------------
// WMMA GEMM:  C[M,N] = act(alpha * A[M,K] @ B[K,N] + bias) + res
// MODEA: 0 = A row-major, 1 = A transposed (A[k*lda+m]), 2 = im2col conv gather
// BH:    0 = B is f32 (weights), 1 = B is f16 (activations)
// PRE:   1 = apply BatchNorm+GELU to the gathered conv input (fused bn_gelu_conv)
// ACT:   0 none, 1 relu, 2 gelu(exact), 3 relu + 0.001 (Performer feature map)
//
// Interior row-major A tiles are staged LDS-side with the CDNA5 async copy
// (global_load_async_to_lds_b128, ASYNCcnt) -- no VGPR round trip.
// ---------------------------------------------------------------------------
template<int MODEA, int BH, int PRE, int ACT>
__global__ __launch_bounds__(128)
void gemm_wmma_k(const _Float16* __restrict__ A,
                 const float*    __restrict__ Bf,
                 const _Float16* __restrict__ Bh,
                 const float*    __restrict__ bias,
                 const _Float16* res,
                 _Float16*       C,
                 int M, int N, int K, int lda, int ldb, int ldc, float alpha,
                 int cs, int cw, int cc, int cp, int cl,
                 const float* __restrict__ bng, const float* __restrict__ bnb,
                 const float* __restrict__ bnm, const float* __restrict__ bnv)
{
    __shared__ _Float16 As[64][40];   // [m][k], rows 80B (16B aligned)
    __shared__ _Float16 Bs[64][40];   // [n][k], fragment-ready contiguous runs

    const int tid  = threadIdx.x;
    const int lane = tid & 31;
    const int wave = tid >> 5;
    const int wm   = wave >> 1;       // 0..1 : 32-row band
    const int wn   = wave & 1;        // 0..1 : 32-col band
    const int m0   = blockIdx.y * 64;
    const int n0   = blockIdx.x * 64;

    const uint64_t abase = (uint64_t)A;

    v8f acc00 = {}, acc01 = {}, acc10 = {}, acc11 = {};

    for (int k0 = 0; k0 < K; k0 += 32) {
        // ---- stage A tile (64 x 32) into LDS ----
        if (MODEA == 0 && (m0 + 64 <= M) && (k0 + 32 <= K)) {
            // CDNA5 async copy: 256 x 16B chunks, 2 per thread, ASYNCcnt-tracked
#pragma unroll
            for (int t = 0; t < 2; t++) {
                int id  = tid + t * 128;
                int r   = id >> 2;          // row 0..63
                int seg = id & 3;           // 8-half segment within row
                unsigned ldsa = (unsigned)(uintptr_t)&As[r][seg * 8];
                unsigned voff = (unsigned)(((size_t)(m0 + r) * lda + k0 + seg * 8)
                                           * sizeof(_Float16));
                asm volatile("global_load_async_to_lds_b128 %0, %1, %2 offset:0"
                             :: "v"(ldsa), "v"(voff), "s"(abase) : "memory");
            }
            asm volatile("s_wait_asynccnt 0x0" ::: "memory");
        } else {
            for (int i = tid; i < 64 * 32; i += 128) {
                int r  = i >> 5, kc = i & 31;
                int gm = m0 + r, gk = k0 + kc;
                float val = 0.f;
                if (gm < M && gk < K) {
                    if (MODEA == 0) {
                        val = (float)A[(size_t)gm * lda + gk];
                    } else if (MODEA == 1) {
                        val = (float)A[(size_t)gk * lda + gm];
                    } else {
                        unsigned w = (unsigned)gk / (unsigned)cc;
                        unsigned c = (unsigned)gk - w * (unsigned)cc;
                        int pos = gm * cs + (int)w - cp;
                        if (pos >= 0 && pos < cl) {
                            float x = (float)A[(size_t)pos * lda + c];
                            if (PRE) {
                                x = (x - bnm[c]) * rsqrtf(bnv[c] + 1e-5f) * bng[c] + bnb[c];
                                x = 0.5f * x * (1.0f + erff(x * 0.70710678118f));
                            }
                            val = x;
                        }
                    }
                }
                As[r][kc] = (_Float16)val;
            }
        }
        // ---- stage B tile (32k x 64n) into LDS, n-major (fragment-ready) ----
        for (int i = tid; i < 32 * 64; i += 128) {
            int nc = i & 63, r = i >> 6;          // coalesced along n
            int gk = k0 + r, gn = n0 + nc;
            float val = 0.f;
            if (gk < K && gn < N)
                val = BH ? (float)Bh[(size_t)gk * ldb + gn]
                         : Bf[(size_t)gk * ldb + gn];
            Bs[nc][r] = (_Float16)val;
        }
        __syncthreads();

        // ---- build wave32 fragments in contiguous k order (element h holds k = 16*(lane/16) + h) ----
        const int ml = lane & 15;
        const int ka = (lane >= 16) ? 16 : 0;   // A: hi-half lanes hold K+16
        const int kb = (lane >= 16) ? 16 : 0;   // B: hi-half lanes hold K+16
        v16h a0, a1, b0, b1;
        const _Float16* bp0 = &Bs[wn * 32 + ml][kb];
        const _Float16* bp1 = &Bs[wn * 32 + 16 + ml][kb];
#pragma unroll
        for (int h = 0; h < 16; h++) {
            int kk = h + ka;
            a0[h] = As[wm * 32 + ml][kk];
            a1[h] = As[wm * 32 + 16 + ml][kk];
            b0[h] = bp0[h];                      // contiguous 32B -> ds_load_b128 x2
            b1[h] = bp1[h];
        }
        acc00 = __builtin_amdgcn_wmma_f32_16x16x32_f16(false, a0, false, b0, (short)0, acc00, false, false);
        acc01 = __builtin_amdgcn_wmma_f32_16x16x32_f16(false, a0, false, b1, (short)0, acc01, false, false);
        acc10 = __builtin_amdgcn_wmma_f32_16x16x32_f16(false, a1, false, b0, (short)0, acc10, false, false);
        acc11 = __builtin_amdgcn_wmma_f32_16x16x32_f16(false, a1, false, b1, (short)0, acc11, false, false);
        __syncthreads();
    }

    // ---- epilogue: alpha, bias, activation, residual ----
    const int nl = lane & 15;
    const int mb = (lane >= 16) ? 8 : 0;
    auto st = [&](float v, int m, int n) {
        if (m >= M || n >= N) return;
        v *= alpha;
        if (bias) v += bias[n];
        if (ACT == 1) v = fmaxf(v, 0.f);
        else if (ACT == 2) v = 0.5f * v * (1.0f + erff(v * 0.70710678118f));
        else if (ACT == 3) v = fmaxf(v, 0.f) + 0.001f;
        if (res) v += (float)res[(size_t)m * ldc + n];
        C[(size_t)m * ldc + n] = (_Float16)v;
    };
#pragma unroll
    for (int e = 0; e < 8; e++) {
        int mr0 = m0 + wm * 32 + e + mb;
        int mr1 = mr0 + 16;
        int nc0 = n0 + wn * 32 + nl;
        int nc1 = nc0 + 16;
        st(acc00[e], mr0, nc0);
        st(acc01[e], mr0, nc1);
        st(acc10[e], mr1, nc0);
        st(acc11[e], mr1, nc1);
    }
}

// ---------------------------------------------------------------------------
// Elementwise / reduction helpers
// ---------------------------------------------------------------------------
__global__ void f32_to_f16_k(const float* in, _Float16* out, int n) {
    int i = blockIdx.x * 256 + threadIdx.x;
    if (i < n) out[i] = (_Float16)in[i];
}

__global__ void copy_cols_k(const _Float16* src, _Float16* dst, int L, int Cs, int ldd) {
    int i = blockIdx.x * 256 + threadIdx.x;
    if (i >= L * Cs) return;
    int l = i / Cs, c = i - l * Cs;
    dst[(size_t)l * ldd + c] = src[i];
}

__global__ void maxpool4_k(const _Float16* in, _Float16* out, int Lo, int C) {
    int i = blockIdx.x * 256 + threadIdx.x;
    if (i >= Lo * C) return;
    int l = i / C, c = i - l * C;
    float m = -1e30f;
    for (int p = 0; p < 4; p++) {
        float v = (float)in[(size_t)(l * 4 + p) * C + c];
        m = v > m ? v : m;
    }
    out[i] = (_Float16)m;
}

__global__ void layernorm_k(const _Float16* x, const float* g, const float* b,
                            _Float16* y, int C) {
    __shared__ float red[256];
    int row = blockIdx.x;
    const _Float16* xr = x + (size_t)row * C;
    float s = 0.f;
    for (int c = threadIdx.x; c < C; c += 256) s += (float)xr[c];
    red[threadIdx.x] = s; __syncthreads();
    for (int st = 128; st > 0; st >>= 1) {
        if (threadIdx.x < st) red[threadIdx.x] += red[threadIdx.x + st];
        __syncthreads();
    }
    float mu = red[0] / C; __syncthreads();
    float s2 = 0.f;
    for (int c = threadIdx.x; c < C; c += 256) {
        float d = (float)xr[c] - mu; s2 += d * d;
    }
    red[threadIdx.x] = s2; __syncthreads();
    for (int st = 128; st > 0; st >>= 1) {
        if (threadIdx.x < st) red[threadIdx.x] += red[threadIdx.x + st];
        __syncthreads();
    }
    float rsig = rsqrtf(red[0] / C + 1e-5f);
    for (int c = threadIdx.x; c < C; c += 256)
        y[(size_t)row * C + c] = (_Float16)(((float)xr[c] - mu) * rsig * g[c] + b[c]);
}

__global__ void rope_k(_Float16* x, int L, int nh, int d) {
    int d2 = d / 2;
    int i = blockIdx.x * 256 + threadIdx.x;
    if (i >= L * nh * d2) return;
    int l = i / (nh * d2);
    int r = i - l * (nh * d2);
    int h = r / d2, j = r - h * d2;
    float inv = powf(10000.0f, -(float)(2 * j) / (float)d);
    float f = (float)l * inv;
    float s = sinf(f), c = cosf(f);
    size_t base = (size_t)l * nh * d + (size_t)h * d;
    float x1 = (float)x[base + j], x2 = (float)x[base + j + d2];
    x[base + j]      = (_Float16)(x1 * c - x2 * s);
    x[base + j + d2] = (_Float16)(x1 * s + x2 * c);
}

__global__ void colsum_k(const _Float16* kp, float* ksum, int L, int C) {
    int c = blockIdx.x * 256 + threadIdx.x;
    if (c >= C) return;
    float s = 0.f;
    for (int l = 0; l < L; l++) s += (float)kp[(size_t)l * C + c];
    ksum[c] = s;
}

__global__ void denom_k(const _Float16* qp, const float* ksum, float* den,
                        int L, int nh, int Mm) {
    int i = blockIdx.x * 256 + threadIdx.x;
    if (i >= L * nh) return;
    int l = i / nh, h = i - l * nh;
    const _Float16* q = qp + (size_t)l * nh * Mm + (size_t)h * Mm;
    const float* ks = ksum + (size_t)h * Mm;
    float s = 0.f;
    for (int m = 0; m < Mm; m++) s += (float)q[m] * ks[m];
    den[i] = s;
}

__global__ void scale_o_k(_Float16* o, const float* den, int L, int nh, int D) {
    int i = blockIdx.x * 256 + threadIdx.x;
    if (i >= L * nh * D) return;
    int l = i / (nh * D);
    int r = i - l * (nh * D);
    int h = r / D;
    o[i] = (_Float16)((float)o[i] / den[l * nh + h]);
}

__global__ void profile_k(const _Float16* y, const float* w, const float* b,
                          float* out, int L, int C) {
    int l = blockIdx.x * 256 + threadIdx.x;
    if (l >= L) return;
    float s = 0.f;
    for (int c = 0; c < C; c++) s += (float)y[(size_t)l * C + c] * w[c];
    s += b[0];
    out[l] = (s > 20.f) ? s : log1pf(expf(s));
}

__global__ void pool_softmax_k(const _Float16* y, const _Float16* wl,
                               float* pooled, int T, int C) {
    int i = blockIdx.x * 256 + threadIdx.x;
    if (i >= T * C) return;
    int t = i / C, c = i - t * C;
    float w[4];
    float mx = -1e30f;
    for (int p = 0; p < 4; p++) {
        w[p] = (float)wl[(size_t)(t * 4 + p) * C + c];
        mx = w[p] > mx ? w[p] : mx;
    }
    float se = 0.f;
    for (int p = 0; p < 4; p++) { w[p] = expf(w[p] - mx); se += w[p]; }
    float acc = 0.f;
    for (int p = 0; p < 4; p++)
        acc += (w[p] / se) * (float)y[(size_t)(t * 4 + p) * C + c];
    pooled[i] = acc;
}

__global__ void peaks_k(const float* pooled, const float* w, const float* b,
                        float* out, int T, int C) {
    int t = blockIdx.x * 256 + threadIdx.x;
    if (t >= T) return;
    float s = 0.f;
    for (int c = 0; c < C; c++) s += pooled[(size_t)t * C + c] * w[c];
    s += b[0];
    out[t] = 1.0f / (1.0f + expf(-s));
}

// ---------------------------------------------------------------------------
// Host orchestration
// ---------------------------------------------------------------------------
struct BNB { const float *g, *b, *m, *v, *k, *kb; };
struct Layer { const float *ln1g,*ln1b,*wq,*wk,*wv,*wo,*bo,*proj,*ln2g,*ln2b,*w1,*b1,*w2,*b2; };

template<int MODEA, int BH, int PRE, int ACT>
static inline void G(hipStream_t st,
                     const _Float16* A, const float* Bf, const _Float16* Bh,
                     const float* bias, const _Float16* res, _Float16* C,
                     int M, int N, int K, int lda, int ldb, int ldc, float alpha,
                     int cs = 1, int cw = 1, int cc = 1, int cp = 0, int cl = 0,
                     const float* g = nullptr, const float* b = nullptr,
                     const float* m = nullptr, const float* v = nullptr)
{
    dim3 grid((N + 63) / 64, (M + 63) / 64);
    gemm_wmma_k<MODEA, BH, PRE, ACT><<<grid, 128, 0, st>>>(
        A, Bf, Bh, bias, res, C, M, N, K, lda, ldb, ldc, alpha,
        cs, cw, cc, cp, cl, g, b, m, v);
}

static inline void* wsal(char*& p, size_t bytes) {
    void* r = (void*)p;
    p += (bytes + 255) & ~(size_t)255;
    return r;
}
static inline int cdiv(int a, int b) { return (a + b - 1) / b; }

extern "C" void kernel_launch(void* const* d_in, const int* in_sizes, int n_in,
                              void* d_out, int out_size, void* d_ws, size_t ws_size,
                              hipStream_t stream)
{
    (void)in_sizes; (void)n_in; (void)out_size; (void)ws_size;
    int pi = 0;
    auto P = [&]() { return (const float*)d_in[pi++]; };
    auto rdBN = [&]() { BNB r; r.g=P(); r.b=P(); r.m=P(); r.v=P(); r.k=P(); r.kb=P(); return r; };

    const float* seq   = P();            // (1,196608,4)
    const float* atac  = P();            // (1,49152,1)
    const float* tfact = P();            // (1,1,1629)
    const float* stem_k = P();           // (15,4,768)
    const float* stem_b = P();
    BNB stem_res = rdBN();
    BNB tower[6]; for (int i = 0; i < 6; i++) tower[i] = rdBN();
    const float* stem_atac_k = P();      // (125,1,32)
    const float* stem_atac_b = P();
    BNB stem_res_atac = rdBN();
    BNB tower_atac[2]; for (int i = 0; i < 2; i++) tower_atac[i] = rdBN();
    const float* tf_w = P();             // (1629,1600)
    const float* tf_b = P();
    Layer lay[6];
    for (int i = 0; i < 6; i++) {
        lay[i].ln1g=P(); lay[i].ln1b=P(); lay[i].wq=P(); lay[i].wk=P(); lay[i].wv=P();
        lay[i].wo=P(); lay[i].bo=P(); lay[i].proj=P(); lay[i].ln2g=P(); lay[i].ln2b=P();
        lay[i].w1=P(); lay[i].b1=P(); lay[i].w2=P(); lay[i].b2=P();
    }
    const float* lnf_g = P();
    const float* lnf_b = P();
    BNB finp = rdBN();
    const float* profile_w = P();
    const float* profile_b = P();
    const float* peaks_pool_w = P();
    const float* peaks_w = P();
    const float* peaks_b = P();

    const int L = 1537, NH = 4, DIM = 400, HID = 1600, Mm = 256;
    const float qscale = 0.22360679775f;   // 400^-0.25
    const float ratio  = 0.0625f;          // 1/sqrt(256)

    // ---- workspace (all activations f16) ----
    char* wp = (char*)d_ws;
    _Float16* seq16  = (_Float16*)wsal(wp, (size_t)196608 * 4 * 2);
    _Float16* atac16 = (_Float16*)wsal(wp, (size_t)49152 * 2);
    _Float16* tf16   = (_Float16*)wsal(wp, (size_t)1629 * 2);
    _Float16* bigA   = (_Float16*)wsal(wp, (size_t)98304 * 768 * 2);
    _Float16* bigB   = (_Float16*)wsal(wp, (size_t)98304 * 768 * 2);
    _Float16* aA     = (_Float16*)wsal(wp, (size_t)24576 * 32 * 2);
    _Float16* aB     = (_Float16*)wsal(wp, (size_t)24576 * 32 * 2);
    _Float16* xbuf   = (_Float16*)wsal(wp, (size_t)L * HID * 2);
    _Float16* hbuf   = (_Float16*)wsal(wp, (size_t)L * HID * 2);
    _Float16* qbuf   = (_Float16*)wsal(wp, (size_t)L * HID * 2);
    _Float16* kbuf   = (_Float16*)wsal(wp, (size_t)L * HID * 2);
    _Float16* vbuf   = (_Float16*)wsal(wp, (size_t)L * HID * 2);
    _Float16* qpb    = (_Float16*)wsal(wp, (size_t)L * NH * Mm * 2);
    _Float16* kpb    = (_Float16*)wsal(wp, (size_t)L * NH * Mm * 2);
    _Float16* kvb    = (_Float16*)wsal(wp, (size_t)NH * Mm * DIM * 2);
    float*    ksumb  = (float*)wsal(wp, (size_t)NH * Mm * 4);
    float*    denb   = (float*)wsal(wp, (size_t)L * NH * 4);
    _Float16* obuf   = (_Float16*)wsal(wp, (size_t)L * HID * 2);
    _Float16* ffnb   = (_Float16*)wsal(wp, (size_t)L * 6400 * 2);
    _Float16* finy   = (_Float16*)wsal(wp, (size_t)896 * 256 * 2);
    _Float16* wlb    = (_Float16*)wsal(wp, (size_t)896 * 256 * 2);
    float*    pooledb= (float*)wsal(wp, (size_t)224 * 256 * 4);

    // ---- input conversion to f16 ----
    f32_to_f16_k<<<cdiv(196608 * 4, 256), 256, 0, stream>>>(seq, seq16, 196608 * 4);
    f32_to_f16_k<<<cdiv(49152, 256), 256, 0, stream>>>(atac, atac16, 49152);
    f32_to_f16_k<<<cdiv(1629, 256), 256, 0, stream>>>(tfact, tf16, 1629);

    // ---- sequence stem: conv W=15 s=2 (pad_left 6) ----
    G<2,0,0,0>(stream, seq16, stem_k, nullptr, stem_b, nullptr, bigA,
               98304, 768, 60, 4, 768, 768, 1.f, 2, 15, 4, 6, 196608);
    // stem residual: x += conv1x1(gelu(bn(x)))  (BN+GELU fused into gather)
    G<2,0,1,0>(stream, bigA, stem_res.k, nullptr, stem_res.kb, bigA, bigB,
               98304, 768, 768, 768, 768, 768, 1.f, 1, 1, 768, 0, 98304,
               stem_res.g, stem_res.b, stem_res.m, stem_res.v);
    _Float16* cur = bigB; _Float16* oth = bigA;

    // ---- conv tower: 6 x (bn+gelu fused conv W=5 s=2, pad_left 1) ----
    int Ls = 98304, cin = 768;
    const int couts[6] = {768, 896, 1024, 1152, 1280, 1536};
    for (int i = 0; i < 6; i++) {
        int Lo = Ls / 2;
        G<2,0,1,0>(stream, cur, tower[i].k, nullptr, tower[i].kb, nullptr, oth,
                   Lo, couts[i], 5 * cin, cin, couts[i], couts[i], 1.f,
                   2, 5, cin, 1, Ls,
                   tower[i].g, tower[i].b, tower[i].m, tower[i].v);
        _Float16* t = cur; cur = oth; oth = t;
        Ls = Lo; cin = couts[i];
    }   // cur: [1536, 1536]

    // ---- ATAC path ----
    G<2,0,0,0>(stream, atac16, stem_atac_k, nullptr, stem_atac_b, nullptr, aA,
               24576, 32, 125, 1, 32, 32, 1.f, 2, 125, 1, 61, 49152);
    G<2,0,1,0>(stream, aA, stem_res_atac.k, nullptr, stem_res_atac.kb, aA, aB,
               24576, 32, 32, 32, 32, 32, 1.f, 1, 1, 32, 0, 24576,
               stem_res_atac.g, stem_res_atac.b, stem_res_atac.m, stem_res_atac.v);
    // tower_atac[0]: conv W=5 s=1 (pad 2), then maxpool4
    G<2,0,1,0>(stream, aB, tower_atac[0].k, nullptr, tower_atac[0].kb, nullptr, aA,
               24576, 32, 160, 32, 32, 32, 1.f, 1, 5, 32, 2, 24576,
               tower_atac[0].g, tower_atac[0].b, tower_atac[0].m, tower_atac[0].v);
    maxpool4_k<<<cdiv(6144 * 32, 256), 256, 0, stream>>>(aA, aB, 6144, 32);
    // tower_atac[1]: 32->64, then maxpool4 -> [1536, 64]
    G<2,0,1,0>(stream, aB, tower_atac[1].k, nullptr, tower_atac[1].kb, nullptr, aA,
               6144, 64, 160, 32, 64, 64, 1.f, 1, 5, 32, 2, 6144,
               tower_atac[1].g, tower_atac[1].b, tower_atac[1].m, tower_atac[1].v);
    maxpool4_k<<<cdiv(1536 * 64, 256), 256, 0, stream>>>(aA, aB, 1536, 64);

    // ---- assemble x[1537, 1600]: tf token row 0, concat(seq, atac) rows 1.. ----
    G<0,0,0,0>(stream, tf16, tf_w, nullptr, tf_b, nullptr, xbuf,
               1, HID, 1629, 1629, HID, HID, 1.f);
    copy_cols_k<<<cdiv(1536 * 1536, 256), 256, 0, stream>>>(cur, xbuf + HID, 1536, 1536, HID);
    copy_cols_k<<<cdiv(1536 * 64, 256), 256, 0, stream>>>(aB, xbuf + HID + 1536, 1536, 64, HID);

    // ---- 6 Performer layers ----
    for (int li = 0; li < 6; li++) {
        const Layer& Ly = lay[li];
        layernorm_k<<<L, 256, 0, stream>>>(xbuf, Ly.ln1g, Ly.ln1b, hbuf, HID);
        G<0,0,0,0>(stream, hbuf, Ly.wq, nullptr, nullptr, nullptr, qbuf,
                   L, HID, HID, HID, HID, HID, qscale);
        G<0,0,0,0>(stream, hbuf, Ly.wk, nullptr, nullptr, nullptr, kbuf,
                   L, HID, HID, HID, HID, HID, qscale);
        G<0,0,0,0>(stream, hbuf, Ly.wv, nullptr, nullptr, nullptr, vbuf,
                   L, HID, HID, HID, HID, HID, 1.f);
        rope_k<<<cdiv(L * NH * (DIM / 2), 256), 256, 0, stream>>>(qbuf, L, NH, DIM);
        rope_k<<<cdiv(L * NH * (DIM / 2), 256), 256, 0, stream>>>(kbuf, L, NH, DIM);
        for (int h = 0; h < NH; h++) {
            G<0,0,0,3>(stream, qbuf + h * DIM, Ly.proj, nullptr, nullptr, nullptr,
                       qpb + h * Mm, L, Mm, DIM, HID, Mm, NH * Mm, ratio);
            G<0,0,0,3>(stream, kbuf + h * DIM, Ly.proj, nullptr, nullptr, nullptr,
                       kpb + h * Mm, L, Mm, DIM, HID, Mm, NH * Mm, ratio);
        }
        colsum_k<<<cdiv(NH * Mm, 256), 256, 0, stream>>>(kpb, ksumb, L, NH * Mm);
        for (int h = 0; h < NH; h++) {
            // kv[h] = kp[h]^T @ v[h] : [256 x 400]
            G<1,1,0,0>(stream, kpb + h * Mm, nullptr, vbuf + h * DIM, nullptr, nullptr,
                       kvb + (size_t)h * Mm * DIM, Mm, DIM, L, NH * Mm, HID, DIM, 1.f);
        }
        denom_k<<<cdiv(L * NH, 256), 256, 0, stream>>>(qpb, ksumb, denb, L, NH, Mm);
        for (int h = 0; h < NH; h++) {
            // o[h] = qp[h] @ kv[h] : [L x 400]
            G<0,1,0,0>(stream, qpb + h * Mm, nullptr, kvb + (size_t)h * Mm * DIM,
                       nullptr, nullptr, obuf + h * DIM,
                       L, DIM, Mm, NH * Mm, DIM, HID, 1.f);
        }
        scale_o_k<<<cdiv(L * HID, 256), 256, 0, stream>>>(obuf, denb, L, NH, DIM);
        // x = x + o @ wo + bo  (in-place: each element read-then-written by owner)
        G<0,0,0,0>(stream, obuf, Ly.wo, nullptr, Ly.bo, xbuf, xbuf,
                   L, HID, HID, HID, HID, HID, 1.f);
        // FFN
        layernorm_k<<<L, 256, 0, stream>>>(xbuf, Ly.ln2g, Ly.ln2b, hbuf, HID);
        G<0,0,0,1>(stream, hbuf, Ly.w1, nullptr, Ly.b1, nullptr, ffnb,
                   L, 4 * HID, HID, HID, 4 * HID, 4 * HID, 1.f);
        G<0,0,0,0>(stream, ffnb, Ly.w2, nullptr, Ly.b2, xbuf, xbuf,
                   L, HID, 4 * HID, 4 * HID, HID, HID, 1.f);
    }

    // ---- final norm, drop token, crop to 896 ----
    layernorm_k<<<L, 256, 0, stream>>>(xbuf, lnf_g, lnf_b, hbuf, HID);
    const _Float16* xc = hbuf + (size_t)(1 + 320) * HID;   // rows 321..1216

    // final_point = bn_gelu_conv (W=1) then outer gelu (ACT=2 in epilogue)
    G<2,0,1,2>(stream, xc, finp.k, nullptr, finp.kb, nullptr, finy,
               896, 256, HID, HID, 256, 256, 1.f, 1, 1, HID, 0, 896,
               finp.g, finp.b, finp.m, finp.v);

    // profile head -> d_out[0..896)
    profile_k<<<cdiv(896, 256), 256, 0, stream>>>(finy, profile_w, profile_b,
                                                  (float*)d_out, 896, 256);
    // peaks head: wlog GEMM, softmax-pool over groups of 4, sigmoid
    G<0,0,0,0>(stream, finy, peaks_pool_w, nullptr, nullptr, nullptr, wlb,
               896, 256, 256, 256, 256, 256, 1.f);
    pool_softmax_k<<<cdiv(224 * 256, 256), 256, 0, stream>>>(finy, wlb, pooledb, 224, 256);
    peaks_k<<<1, 256, 0, stream>>>(pooledb, peaks_w, peaks_b,
                                   (float*)d_out + 896, 224, 256);
}
